// GAT_78752520340206
// MI455X (gfx1250) — matrix operand order repaired and checked by reading, not compiled, not run on hardware
//
#include <hip/hip_runtime.h>
#include <hip/hip_bf16.h>
#include <cstdint>
#include <cstddef>

// ---------------------------------------------------------------------------
// Types for CDNA5 WMMA
// ---------------------------------------------------------------------------
typedef __attribute__((ext_vector_type(16))) __bf16 v16bf;
typedef __attribute__((ext_vector_type(8)))  float  v8f;
typedef __attribute__((ext_vector_type(8)))  int    v8i;

#define NEG_SLOPE 0.2f

// ---------------------------------------------------------------------------
// Helpers
// ---------------------------------------------------------------------------
static __device__ __forceinline__ unsigned short f32_to_bf16_rne(float f) {
  union { float f; unsigned u; } c; c.f = f;
  unsigned u = c.u;
  unsigned lsb = (u >> 16) & 1u;
  u += 0x7fffu + lsb;
  return (unsigned short)(u >> 16);
}

// Float atomic max via the sign-split integer idiom (init must be < all vals)
static __device__ __forceinline__ void atomicMaxF32(float* addr, float value) {
  if (value >= 0.0f)
    atomicMax((int*)addr, __float_as_int(value));
  else
    atomicMin((unsigned int*)addr, (unsigned int)__float_as_int(value));
}

// ---------------------------------------------------------------------------
// Elementwise / conversion kernels
// ---------------------------------------------------------------------------
__global__ void cvt_f32_bf16(const float* __restrict__ src,
                             unsigned short* __restrict__ dst, int n) {
  int i = blockIdx.x * blockDim.x + threadIdx.x;
  if (i < n) dst[i] = f32_to_bf16_rne(src[i]);
}

__global__ void fill_f32(float* __restrict__ p, float v, int n) {
  int i = blockIdx.x * blockDim.x + threadIdx.x;
  if (i < n) p[i] = v;
}

// ---------------------------------------------------------------------------
// bf16 WMMA GEMM:  C[M,Nn] = A[M,K] * B[Nn,K]^T   (both bf16 row-major)
// One wave computes a 16x64 strip of C (4 N-tiles sharing one A fragment):
// 4x v_wmma_f32_16x16x32_bf16 per K-step. Fragment layouts per ISA 7.12.2.
// Requires M%16==0, Nn%64==0, K%32==0 (true for both GEMMs here).
// ---------------------------------------------------------------------------
__global__ __launch_bounds__(256) void gemm_bf16_wmma4(
    const unsigned short* __restrict__ A, const unsigned short* __restrict__ B,
    float* __restrict__ C, int M, int Nn, int K) {
  int wave = (int)((blockIdx.x * blockDim.x + threadIdx.x) >> 5);
  int lane = threadIdx.x & 31;
  int nblk_n = Nn >> 6;                  // 64-wide N blocks
  int nwaves = (M >> 4) * nblk_n;
  if (wave >= nwaves) return;
  int tm = wave / nblk_n;
  int bn = wave - tm * nblk_n;
  int half = lane >> 4;
  int l15  = lane & 15;

  const unsigned short* arow = A + (size_t)(tm * 16 + l15) * K;
  const unsigned short* brow0 = B + (size_t)(bn * 64 + l15) * K;
  const size_t bstride = (size_t)16 * K;   // next 16-column tile of B

  v8f acc0 = {}, acc1 = {}, acc2 = {}, acc3 = {};
  for (int k0 = 0; k0 < K; k0 += 32) {
    v8i ai;
#pragma unroll
    for (int j = 0; j < 8; ++j) {
      // A 16x32 bf16 fragment: lanes 0-15 rows M with K pairs {0..7,16..23};
      // lanes 16-31 same rows, K pairs +8.
      int ka = k0 + 2 * j + (j < 4 ? 0 : 8) + half * 8;
      ai[j] = *(const int*)(arow + ka);
    }
    v16bf av = __builtin_bit_cast(v16bf, ai);

#pragma unroll
    for (int t = 0; t < 4; ++t) {
      v8i bi;
      const unsigned short* brow = brow0 + (size_t)t * bstride;
#pragma unroll
      for (int j = 0; j < 8; ++j) {
        // B 32x16 fragment: lanes 0-15 cols N, K {0..7,16..23}; lanes 16-31 K +8
        int kb = k0 + 2 * j + (j < 4 ? 0 : 8) + half * 8;
        bi[j] = *(const int*)(brow + kb);
      }
      v16bf bv = __builtin_bit_cast(v16bf, bi);
      v8f* accp = (t == 0) ? &acc0 : (t == 1) ? &acc1 : (t == 2) ? &acc2 : &acc3;
      *accp = __builtin_amdgcn_wmma_f32_16x16x32_bf16(
          false, av, false, bv, (short)0, *accp, false, false);
    }
  }

  // C/D layout: VGPR r -> row (r + 8*half), col l15 within each 16x16 tile
  float* cbase = C + (size_t)(tm * 16 + 8 * half) * Nn + bn * 64 + l15;
#pragma unroll
  for (int t = 0; t < 4; ++t) {
    const v8f& acc = (t == 0) ? acc0 : (t == 1) ? acc1 : (t == 2) ? acc2 : acc3;
#pragma unroll
    for (int r = 0; r < 8; ++r)
      cbase[(size_t)r * Nn + t * 16] = acc[r];
  }
}

// ---------------------------------------------------------------------------
// a_src[n,h] = sum_c h[n,h,c]*att_src[h,c];  a_dst likewise
// ---------------------------------------------------------------------------
__global__ void attn_scores(const float* __restrict__ h,
                            const float* __restrict__ att_s,
                            const float* __restrict__ att_d,
                            float* __restrict__ as, float* __restrict__ ad,
                            int N, int H, int C) {
  int idx = blockIdx.x * blockDim.x + threadIdx.x;
  if (idx >= N * H) return;
  int n = idx / H, hh = idx - n * H;
  const float* row = h + (size_t)n * H * C + (size_t)hh * C;
  const float* wsp = att_s + (size_t)hh * C;
  const float* wdp = att_d + (size_t)hh * C;
  float ss = 0.f, dd = 0.f;
  for (int c = 0; c < C; ++c) { float v = row[c]; ss += v * wsp[c]; dd += v * wdp[c]; }
  as[idx] = ss; ad[idx] = dd;
}

// ---------------------------------------------------------------------------
// Edge softmax passes (edges = E real edges + N self loops)
// ---------------------------------------------------------------------------
__global__ void edge_max(const int* __restrict__ ei, int E, int N, int H,
                         const float* __restrict__ asrc,
                         const float* __restrict__ adst,
                         float* __restrict__ m) {
  int idx = blockIdx.x * blockDim.x + threadIdx.x;
  int total = (E + N) * H;
  if (idx >= total) return;
  int e = idx / H, h = idx - e * H;
  int s = (e < E) ? ei[e]     : (e - E);
  int d = (e < E) ? ei[E + e] : (e - E);
  float v = asrc[s * H + h] + adst[d * H + h];
  v = (v > 0.f) ? v : NEG_SLOPE * v;
  atomicMaxF32(&m[d * H + h], v);
}

__global__ void edge_expsum(const int* __restrict__ ei, int E, int N, int H,
                            const float* __restrict__ asrc,
                            const float* __restrict__ adst,
                            const float* __restrict__ m,
                            float* __restrict__ den) {
  int idx = blockIdx.x * blockDim.x + threadIdx.x;
  int total = (E + N) * H;
  if (idx >= total) return;
  int e = idx / H, h = idx - e * H;
  int s = (e < E) ? ei[e]     : (e - E);
  int d = (e < E) ? ei[E + e] : (e - E);
  float v = asrc[s * H + h] + adst[d * H + h];
  v = (v > 0.f) ? v : NEG_SLOPE * v;
  atomicAdd(&den[d * H + h], __expf(v - m[d * H + h]));
}

// alpha[e,h] = exp(e_score - m[dst,h]) / (den[dst,h] + 1e-16)
__global__ void edge_alpha(const int* __restrict__ ei, int E, int N, int H,
                           const float* __restrict__ asrc,
                           const float* __restrict__ adst,
                           const float* __restrict__ m,
                           const float* __restrict__ den,
                           float* __restrict__ alpha) {
  int idx = blockIdx.x * blockDim.x + threadIdx.x;
  int total = (E + N) * H;
  if (idx >= total) return;
  int e = idx / H, h = idx - e * H;
  int s = (e < E) ? ei[e]     : (e - E);
  int d = (e < E) ? ei[E + e] : (e - E);
  float v = asrc[s * H + h] + adst[d * H + h];
  v = (v > 0.f) ? v : NEG_SLOPE * v;
  alpha[idx] = __expf(v - m[d * H + h]) / (den[d * H + h] + 1e-16f);
}

// agg[dst, :] += h[src, :] * alpha ; 4 channels per thread (float4 gather)
__global__ void edge_agg4(const int* __restrict__ ei, int E, int N,
                          int H, int C,
                          const float* __restrict__ alpha,
                          const float* __restrict__ hsrc,
                          float* __restrict__ agg) {
  int D = H * C;
  int q = D >> 2;
  long long idx = (long long)blockIdx.x * blockDim.x + threadIdx.x;
  long long total = (long long)(E + N) * q;
  if (idx >= total) return;
  int e  = (int)(idx / q);
  int c4 = (int)(idx - (long long)e * q) << 2;   // starting channel (C%4==0)
  int h  = c4 / C;
  int s = (e < E) ? ei[e]     : (e - E);
  int d = (e < E) ? ei[E + e] : (e - E);
  float a = alpha[e * H + h];
  float4 hv = *(const float4*)(hsrc + (size_t)s * D + c4);
  float* dst = agg + (size_t)d * D + c4;
  atomicAdd(dst + 0, hv.x * a);
  atomicAdd(dst + 1, hv.y * a);
  atomicAdd(dst + 2, hv.z * a);
  atomicAdd(dst + 3, hv.w * a);
}

// ---------------------------------------------------------------------------
// Bias + ReLU (emit bf16 for next GEMM, or f32 in-place)
// ---------------------------------------------------------------------------
__global__ void bias_relu_bf16(const float* __restrict__ acc,
                               const float* __restrict__ bias,
                               unsigned short* __restrict__ out,
                               int total, int D) {
  int i = blockIdx.x * blockDim.x + threadIdx.x;
  if (i >= total) return;
  float v = acc[i] + bias[i % D];
  v = (v > 0.f) ? v : 0.f;
  out[i] = f32_to_bf16_rne(v);
}

__global__ void bias_relu_f32(float* __restrict__ acc,
                              const float* __restrict__ bias,
                              int total, int D) {
  int i = blockIdx.x * blockDim.x + threadIdx.x;
  if (i >= total) return;
  float v = acc[i] + bias[i % D];
  acc[i] = (v > 0.f) ? v : 0.f;
}

// ---------------------------------------------------------------------------
// Attention pooling
// ---------------------------------------------------------------------------
__global__ void pool_scale(const float* __restrict__ h,
                           const float* __restrict__ w_attn,
                           const float* __restrict__ b_attn,
                           const float* __restrict__ w_mask,
                           const float* __restrict__ b_mask,
                           float* __restrict__ s, int N, int D) {
  int n = blockIdx.x * blockDim.x + threadIdx.x;
  if (n >= N) return;
  float a = b_attn[0], mm = b_mask[0];
  const float* row = h + (size_t)n * D;
  for (int c = 0; c < D; ++c) { a += row[c] * w_attn[c]; mm += row[c] * w_mask[c]; }
  s[n] = a * (1.f / (1.f + __expf(-mm)));
}

__global__ void pool_scatter4(const float* __restrict__ h,
                              const float* __restrict__ s,
                              const int* __restrict__ batch,
                              float* __restrict__ pooled, int N, int D) {
  int q = D >> 2;
  long long idx = (long long)blockIdx.x * blockDim.x + threadIdx.x;
  if (idx >= (long long)N * q) return;
  int n  = (int)(idx / q);
  int c4 = (int)(idx - (long long)n * q) << 2;
  float sc = s[n];
  float4 hv = *(const float4*)(h + (size_t)n * D + c4);
  float* dst = pooled + (size_t)batch[n] * D + c4;
  atomicAdd(dst + 0, hv.x * sc);
  atomicAdd(dst + 1, hv.y * sc);
  atomicAdd(dst + 2, hv.z * sc);
  atomicAdd(dst + 3, hv.w * sc);
}

__global__ void final_linear(const float* __restrict__ pooled,
                             const float* __restrict__ W,
                             const float* __restrict__ b,
                             float* __restrict__ out, int G, int D, int O) {
  int idx = blockIdx.x * blockDim.x + threadIdx.x;
  if (idx >= G * O) return;
  int g = idx / O, o = idx - g * O;
  float acc = b[o];
  const float* prow = pooled + (size_t)g * D;
  const float* wrow = W + (size_t)o * D;
  for (int c = 0; c < D; ++c) acc += prow[c] * wrow[c];
  out[idx] = acc;
}

// ---------------------------------------------------------------------------
// Host launcher
// ---------------------------------------------------------------------------
#define CEILDIV(a, b) (((a) + (b) - 1) / (b))

extern "C" void kernel_launch(void* const* d_in, const int* in_sizes, int n_in,
                              void* d_out, int out_size, void* d_ws, size_t ws_size,
                              hipStream_t stream) {
  const float* x      = (const float*)d_in[0];
  const int*   ei     = (const int*)  d_in[1];
  const int*   batch  = (const int*)  d_in[2];
  const float* W1     = (const float*)d_in[3];
  const float* as1w   = (const float*)d_in[4];
  const float* ad1w   = (const float*)d_in[5];
  const float* b1     = (const float*)d_in[6];
  const float* W2     = (const float*)d_in[7];
  const float* as2w   = (const float*)d_in[8];
  const float* ad2w   = (const float*)d_in[9];
  const float* b2     = (const float*)d_in[10];
  const float* w_attn = (const float*)d_in[11];
  const float* b_attn = (const float*)d_in[12];
  const float* w_mask = (const float*)d_in[13];
  const float* b_mask = (const float*)d_in[14];
  const float* W_out  = (const float*)d_in[15];
  const float* b_out  = (const float*)d_in[16];

  const int N   = in_sizes[2];            // 20000
  const int E   = in_sizes[1] / 2;        // 320000
  const int Fin = in_sizes[0] / N;        // 128
  const int D1  = in_sizes[6];            // 512
  const int H1  = 8;                      // reference constant
  const int C1  = D1 / H1;                // 64
  const int D2  = in_sizes[10];           // 128
  const int G   = 256;                    // NUM_GRAPHS
  const int OUT = in_sizes[16];           // 2

  // ---- workspace layout ----
  char* ws = (char*)d_ws;
  size_t off = 0;
  auto alloc = [&](size_t bytes) -> size_t {
    size_t o = off;
    off = (off + bytes + 255) & ~(size_t)255;
    return o;
  };
  size_t o_h1    = alloc((size_t)N * D1 * 4);        // GEMM1 out (f32)
  size_t o_agg1  = alloc((size_t)N * D1 * 4);        // layer-1 aggregation
  size_t o_xbf   = alloc((size_t)N * Fin * 2);       // x bf16
  size_t o_w1bf  = alloc((size_t)D1 * Fin * 2);
  size_t o_as1   = alloc((size_t)N * H1 * 4);
  size_t o_ad1   = alloc((size_t)N * H1 * 4);
  size_t o_m1    = alloc((size_t)N * H1 * 4);
  size_t o_den1  = alloc((size_t)N * H1 * 4);
  size_t o_al1   = alloc((size_t)(E + N) * H1 * 4);  // alpha layer 1
  size_t o_h1rbf = alloc((size_t)N * D1 * 2);        // relu(agg1+b1) bf16
  size_t o_w2bf  = alloc((size_t)D2 * D1 * 2);
  size_t o_h2    = alloc((size_t)N * D2 * 4);        // GEMM2 out (f32)
  size_t o_agg2  = alloc((size_t)N * D2 * 4);
  size_t o_as2   = alloc((size_t)N * 4);
  size_t o_ad2   = alloc((size_t)N * 4);
  size_t o_m2    = alloc((size_t)N * 4);
  size_t o_den2  = alloc((size_t)N * 4);
  size_t o_al2   = alloc((size_t)(E + N) * 4);       // alpha layer 2
  size_t o_s     = alloc((size_t)N * 4);
  size_t o_pool  = alloc((size_t)G * D2 * 4);
  (void)ws_size; (void)n_in; (void)out_size;

  float*          h1    = (float*)(ws + o_h1);
  float*          agg1  = (float*)(ws + o_agg1);
  unsigned short* xbf   = (unsigned short*)(ws + o_xbf);
  unsigned short* w1bf  = (unsigned short*)(ws + o_w1bf);
  float*          as1   = (float*)(ws + o_as1);
  float*          ad1   = (float*)(ws + o_ad1);
  float*          m1    = (float*)(ws + o_m1);
  float*          den1  = (float*)(ws + o_den1);
  float*          al1   = (float*)(ws + o_al1);
  unsigned short* h1rbf = (unsigned short*)(ws + o_h1rbf);
  unsigned short* w2bf  = (unsigned short*)(ws + o_w2bf);
  float*          h2    = (float*)(ws + o_h2);
  float*          agg2  = (float*)(ws + o_agg2);
  float*          as2   = (float*)(ws + o_as2);
  float*          ad2   = (float*)(ws + o_ad2);
  float*          m2    = (float*)(ws + o_m2);
  float*          den2  = (float*)(ws + o_den2);
  float*          al2   = (float*)(ws + o_al2);
  float*          sbuf  = (float*)(ws + o_s);
  float*          pooled= (float*)(ws + o_pool);

  const int T = 256;

  // ---- init accumulators ----
  hipMemsetAsync(agg1,  0, (size_t)N * D1 * 4, stream);
  hipMemsetAsync(den1,  0, (size_t)N * H1 * 4, stream);
  hipMemsetAsync(agg2,  0, (size_t)N * D2 * 4, stream);
  hipMemsetAsync(den2,  0, (size_t)N * 4,      stream);
  hipMemsetAsync(pooled,0, (size_t)G * D2 * 4, stream);
  fill_f32<<<CEILDIV(N * H1, T), T, 0, stream>>>(m1, -3.0e38f, N * H1);
  fill_f32<<<CEILDIV(N, T), T, 0, stream>>>(m2, -3.0e38f, N);

  // ---- precision conversions ----
  cvt_f32_bf16<<<CEILDIV(N * Fin, T), T, 0, stream>>>(x,  xbf,  N * Fin);
  cvt_f32_bf16<<<CEILDIV(D1 * Fin, T), T, 0, stream>>>(W1, w1bf, D1 * Fin);
  cvt_f32_bf16<<<CEILDIV(D2 * D1, T), T, 0, stream>>>(W2, w2bf, D2 * D1);

  // ---- layer 1: GEMM h1 = x @ W1^T ----
  {
    int nwaves = (N / 16) * (D1 / 64);
    gemm_bf16_wmma4<<<CEILDIV(nwaves * 32, T), T, 0, stream>>>(xbf, w1bf, h1, N, D1, Fin);
  }
  attn_scores<<<CEILDIV(N * H1, T), T, 0, stream>>>(h1, as1w, ad1w, as1, ad1, N, H1, C1);
  edge_max   <<<CEILDIV((E + N) * H1, T), T, 0, stream>>>(ei, E, N, H1, as1, ad1, m1);
  edge_expsum<<<CEILDIV((E + N) * H1, T), T, 0, stream>>>(ei, E, N, H1, as1, ad1, m1, den1);
  edge_alpha <<<CEILDIV((E + N) * H1, T), T, 0, stream>>>(ei, E, N, H1, as1, ad1, m1, den1, al1);
  {
    long long total = (long long)(E + N) * (D1 / 4);
    edge_agg4<<<(unsigned)CEILDIV(total, (long long)T), T, 0, stream>>>(
        ei, E, N, H1, C1, al1, h1, agg1);
  }
  bias_relu_bf16<<<CEILDIV(N * D1, T), T, 0, stream>>>(agg1, b1, h1rbf, N * D1, D1);

  // ---- layer 2: GEMM h2 = relu1 @ W2^T ----
  {
    int nwaves = (N / 16) * (D2 / 64);
    gemm_bf16_wmma4<<<CEILDIV(nwaves * 32, T), T, 0, stream>>>(h1rbf, w2bf, h2, N, D2, D1);
  }
  attn_scores<<<CEILDIV(N, T), T, 0, stream>>>(h2, as2w, ad2w, as2, ad2, N, 1, D2);
  edge_max   <<<CEILDIV(E + N, T), T, 0, stream>>>(ei, E, N, 1, as2, ad2, m2);
  edge_expsum<<<CEILDIV(E + N, T), T, 0, stream>>>(ei, E, N, 1, as2, ad2, m2, den2);
  edge_alpha <<<CEILDIV(E + N, T), T, 0, stream>>>(ei, E, N, 1, as2, ad2, m2, den2, al2);
  {
    long long total = (long long)(E + N) * (D2 / 4);
    edge_agg4<<<(unsigned)CEILDIV(total, (long long)T), T, 0, stream>>>(
        ei, E, N, 1, D2, al2, h2, agg2);
  }
  bias_relu_f32<<<CEILDIV(N * D2, T), T, 0, stream>>>(agg2, b2, N * D2, D2);

  // ---- attention pooling ----
  pool_scale<<<CEILDIV(N, T), T, 0, stream>>>(agg2, w_attn, b_attn, w_mask, b_mask, sbuf, N, D2);
  pool_scatter4<<<CEILDIV(N * (D2 / 4), T), T, 0, stream>>>(agg2, sbuf, batch, pooled, N, D2);

  // ---- final linear ----
  final_linear<<<CEILDIV(G * OUT, T), T, 0, stream>>>(pooled, W_out, b_out, (float*)d_out, G, D2, OUT);
}
